// SynthesisBlock_78039555768979
// MI455X (gfx1250) — matrix operand order repaired and checked by reading, not compiled, not run on hardware
//
#include <hip/hip_runtime.h>

typedef float v8f __attribute__((ext_vector_type(8)));
typedef __bf16 v16bf __attribute__((ext_vector_type(16)));
typedef unsigned int uint4v __attribute__((ext_vector_type(4)));
typedef unsigned int uint2v __attribute__((ext_vector_type(2)));

#define IN_CH 256
#define OUT_CH 256
#define RES 128
#define LATENT 512
#define A_PITCH 40   /* halves per 32-ci row (64B data + 16B pad), 80B = 16B-aligned, conflict-free */
#define B_PITCH 40

__device__ __forceinline__ unsigned short f2bf(float f) {
  unsigned int u = __float_as_uint(f);
  u += 0x7FFFu + ((u >> 16) & 1u);   // round-to-nearest-even
  return (unsigned short)(u >> 16);
}

// ---- kernel 1: pack conv_w -> bf16 [tap][co][ci], and w_sq[o][i] = sum_t w^2 ----
__global__ __launch_bounds__(256) void pack_kernel(const float* __restrict__ conv_w,
                                                   unsigned short* __restrict__ wpk,
                                                   float* __restrict__ wsq) {
  int g = blockIdx.x * 256 + threadIdx.x;          // g = co*256 + ci
  float s = 0.f;
#pragma unroll
  for (int t = 0; t < 9; ++t) {
    float f = conv_w[g * 9 + t];
    s += f * f;
    wpk[t * 65536 + g] = f2bf(f);                  // [t][co][ci]
  }
  wsq[g] = s;
}

// ---- kernel 2: style[b][i] = (latent[b] . aff_w[i]) * sqrt(1/512) + aff_b[i] ----
__global__ __launch_bounds__(256) void style_kernel(const float* __restrict__ latent,
                                                    const float* __restrict__ aff_w,
                                                    const float* __restrict__ aff_b,
                                                    float* __restrict__ style) {
  int b = blockIdx.x, i = threadIdx.x;
  const float* lr = latent + b * LATENT;
  const float* wr = aff_w + (size_t)i * LATENT;
  float acc = 0.f;
  for (int l = 0; l < LATENT; ++l) acc += lr[l] * wr[l];
  style[b * 256 + i] = acc * 0.04419417382415922f + aff_b[i];
}

// ---- kernel 3: demod[b][o] = rsqrt(sum_i wsq[o][i]*style[b][i]^2 + 1e-8) ----
__global__ __launch_bounds__(256) void demod_kernel(const float* __restrict__ wsq,
                                                    const float* __restrict__ style,
                                                    float* __restrict__ demod) {
  __shared__ float s2[256];
  int b = blockIdx.x, o = threadIdx.x;
  float sv = style[b * 256 + o];
  s2[o] = sv * sv;
  __syncthreads();
  const float* wr = wsq + (size_t)o * 256;
  float acc = 0.f;
  for (int i = 0; i < 256; ++i) acc += wr[i] * s2[i];
  demod[b * 256 + o] = rsqrtf(acc + 1e-8f);
}

// ---- kernel 4: implicit-GEMM modulated conv via WMMA bf16 ----
// block: 256 threads (8 waves), tile = 64 cout x (8x16) pixels, one batch
// waves arranged 2 (cout) x 4 (pixels); each wave: 2x2 fragments of 16x16
__global__ __launch_bounds__(256) void conv_kernel(
    const float* __restrict__ content, const float* __restrict__ noise,
    const float* __restrict__ bias, const float* __restrict__ noise_w,
    const unsigned short* __restrict__ wpk, const float* __restrict__ style,
    const float* __restrict__ demod, float* __restrict__ out) {
  __shared__ __align__(16) unsigned short As[9 * 64 * A_PITCH];  // [tap][co 64][ci 32]
  __shared__ __align__(16) unsigned short Bs[180 * B_PITCH];     // [halo pix 10x18][ci 32]
  __shared__ float s_style[256];

  const int tid = threadIdx.x;
  const int b   = blockIdx.z;
  const int co0 = blockIdx.y * 64;
  const int ph0 = (blockIdx.x >> 3) * 8;    // 16 row tiles
  const int pw0 = (blockIdx.x & 7) * 16;    // 8 col tiles

  const int lane = tid & 31;
  const int wave = tid >> 5;
  const int wm = wave & 1;                  // cout half (32 couts)
  const int wn = wave >> 1;                 // pixel quarter (32 pixels)
  const int m_l  = lane & 15;
  const int kb_a = (lane < 16) ? 0 : 8;     // ISA 16-bit A layout: K base per half-wave
  const int kb_b = (lane < 16) ? 0 : 8;     // ISA 16-bit B layout: K base per half-wave

  s_style[tid] = style[b * 256 + tid];

  v8f acc[2][2] = {};

  for (int c = 0; c < 8; ++c) {             // 8 chunks of 32 input channels
    const int ci0 = c * 32;
    __syncthreads();
    // --- stage modulated content halo (10x18) as bf16, layout [pix][ci] ---
    for (int i = tid; i < 32 * 180; i += 256) {
      int ci  = i / 180;
      int p   = i - ci * 180;
      int row = p / 18;
      int col = p - row * 18;
      int y = ph0 + row - 1;
      int x = pw0 + col - 1;
      float v = 0.f;
      if ((unsigned)y < RES && (unsigned)x < RES)
        v = content[(((size_t)(b * IN_CH + ci0 + ci)) * RES + y) * RES + x] * s_style[ci0 + ci];
      Bs[p * B_PITCH + ci] = f2bf(v);
    }
    // --- prefetch next channel chunk (global_prefetch_b8) ---
    if (c < 7 && tid < 8) {
      int y = ph0 + tid;
      __builtin_prefetch(&content[(((size_t)(b * IN_CH + ci0 + 32 + (tid & 3))) * RES + y) * RES + pw0], 0, 1);
    }
    // --- stage weights: 9 taps x 64 co x 32 ci, 8-byte copies ---
    for (int i = tid; i < 9 * 64 * 8; i += 256) {
      int t   = i >> 9;
      int rem = i & 511;
      int co  = rem >> 3;
      int cg  = rem & 7;
      uint2v w = *(const uint2v*)(wpk + (size_t)t * 65536 + (size_t)(co0 + co) * 256 + ci0 + cg * 4);
      *(uint2v*)(&As[(t * 64 + co) * A_PITCH + cg * 4]) = w;
    }
    __syncthreads();

#pragma unroll
    for (int t = 0; t < 9; ++t) {
      const int th = t / 3;
      const int tw = t - th * 3;
      union { uint4v q[2]; v16bf v; } a[2], bm[2];
#pragma unroll
      for (int mi = 0; mi < 2; ++mi) {
        // A fragment: lane m in [0,16); halves 0..7 = K kb_a..kb_a+7, 8..15 = K kb_a+16..kb_a+23
        const unsigned short* pA = &As[(t * 64 + wm * 32 + mi * 16 + m_l) * A_PITCH + kb_a];
        a[mi].q[0] = *(const uint4v*)(pA);
        a[mi].q[1] = *(const uint4v*)(pA + 16);
      }
#pragma unroll
      for (int ni = 0; ni < 2; ++ni) {
        // B fragment: N = lane&15 within 16-pixel row; halves 0..7 = K kb_b..kb_b+7, 8..15 = K kb_b+16..kb_b+23
        int n = wn * 32 + ni * 16 + m_l;
        int p = ((n >> 4) + th) * 18 + (n & 15) + tw;   // shifted halo pixel for this tap
        const unsigned short* pB = &Bs[p * B_PITCH + kb_b];
        bm[ni].q[0] = *(const uint4v*)(pB);
        bm[ni].q[1] = *(const uint4v*)(pB + 16);
      }
#pragma unroll
      for (int mi = 0; mi < 2; ++mi)
#pragma unroll
        for (int ni = 0; ni < 2; ++ni)
          acc[mi][ni] = __builtin_amdgcn_wmma_f32_16x16x32_bf16(
              false, a[mi].v, false, bm[ni].v, (short)0, acc[mi][ni], false, false);
    }
  }

  // --- fused epilogue: demod, noise, bias, leaky_relu(0.2) * sqrt(2) ---
  const float nw = noise_w[0];
#pragma unroll
  for (int ni = 0; ni < 2; ++ni) {
    int n = wn * 32 + ni * 16 + m_l;
    int y = ph0 + (n >> 4);
    int x = pw0 + (n & 15);
    float nz = nw * noise[(size_t)b * (RES * RES) + y * RES + x];
#pragma unroll
    for (int mi = 0; mi < 2; ++mi) {
      union { v8f v; float f[8]; } u;
      u.v = acc[mi][ni];
#pragma unroll
      for (int r = 0; r < 8; ++r) {
        int co = co0 + wm * 32 + mi * 16 + ((lane < 16) ? r : r + 8);  // C/D layout: M = r (+8 upper half)
        float v = u.f[r] * demod[b * 256 + co] + nz + bias[co];
        v = (v > 0.f ? v : 0.2f * v) * 1.4142135623730951f;
        out[(((size_t)(b * OUT_CH + co)) * RES + y) * RES + x] = v;
      }
    }
  }
}

extern "C" void kernel_launch(void* const* d_in, const int* in_sizes, int n_in,
                              void* d_out, int out_size, void* d_ws, size_t ws_size,
                              hipStream_t stream) {
  const float* content = (const float*)d_in[0];
  const float* latent  = (const float*)d_in[1];
  const float* noise   = (const float*)d_in[2];
  const float* aff_w   = (const float*)d_in[3];
  const float* aff_b   = (const float*)d_in[4];
  const float* conv_w  = (const float*)d_in[5];
  const float* bias    = (const float*)d_in[6];
  const float* noise_w = (const float*)d_in[7];
  float* out = (float*)d_out;

  // workspace layout (~1.47 MB)
  unsigned short* wpk = (unsigned short*)d_ws;                   // 9*256*256 bf16
  float* wsq   = (float*)((char*)d_ws + 1179648);                // 256*256 f32
  float* style = (float*)((char*)d_ws + 1441792);                // 16*256 f32
  float* demod = (float*)((char*)d_ws + 1458176);                // 16*256 f32

  pack_kernel <<<256, 256, 0, stream>>>(conv_w, wpk, wsq);
  style_kernel<<<16, 256, 0, stream>>>(latent, aff_w, aff_b, style);
  demod_kernel<<<16, 256, 0, stream>>>(wsq, style, demod);
  conv_kernel <<<dim3(128, 4, 16), 256, 0, stream>>>(content, noise, bias, noise_w,
                                                     wpk, style, demod, out);
}
